// MultiHeadAttention_23235773072153
// MI455X (gfx1250) — matrix operand order repaired and checked by reading, not compiled, not run on hardware
//
#include <hip/hip_runtime.h>
#include <hip/hip_bf16.h>
#include <math.h>

typedef __attribute__((ext_vector_type(16))) __bf16 v16bf;
typedef __attribute__((ext_vector_type(8)))  __bf16 v8bf;
typedef __attribute__((ext_vector_type(8)))  float  v8f;

#define D_MODEL 1024
#define S_LEN   2048
#define NHEAD   16
#define DK      64

static __device__ __forceinline__ v16bf join16(v8bf lo, v8bf hi) {
  v16bf r;
#pragma unroll
  for (int i = 0; i < 8; ++i) { r[i] = lo[i]; r[i + 8] = hi[i]; }
  return r;
}

// ---------------------------------------------------------------------------
// Projection GEMM: C[M x 1024] = A[M x 1024] @ W[1024 x 1024] + bias
// MODE 0: store bf16 into head layout   [B,H,S,64]   (Q, K)
// MODE 1: store bf16 into head-T layout [B,H,64,S]   (V)
// MODE 2: store fp32 row-major with ReLU              (final out)
// Block: 128 threads (4 waves). Block tile 64x64; wave tile 16x64.
// Double-buffered LDS; next K-tile global loads issued before current WMMAs.
// ---------------------------------------------------------------------------
template <typename AT, int MODE>
__global__ __launch_bounds__(128) void proj_gemm(const AT* __restrict__ A,
                                                 const float* __restrict__ W,
                                                 const float* __restrict__ bias,
                                                 void* __restrict__ out) {
  __shared__ __align__(16) __bf16 lds_a[2][64 * 32];   // [m][k]
  __shared__ __align__(16) __bf16 lds_w[2][64 * 32];   // transposed [n][k]

  const int t    = threadIdx.x;
  const int lane = t & 31;
  const int wv   = t >> 5;
  const bool lo  = lane < 16;
  const int l16  = lane & 15;
  const int n0   = blockIdx.x * 64;
  const int m0   = blockIdx.y * 64;

  float ra[16], rw[16];
  // Prologue: stage K-tile 0 into LDS buffer 0.
#pragma unroll
  for (int i = 0; i < 16; ++i) {
    int e = t + i * 128;
    ra[i] = (float)A[(size_t)(m0 + (e >> 5)) * D_MODEL + (e & 31)];
    rw[i] = W[(size_t)(e >> 6) * D_MODEL + n0 + (e & 63)];
  }
#pragma unroll
  for (int i = 0; i < 16; ++i) {
    int e = t + i * 128;
    lds_a[0][(e >> 5) * 32 + (e & 31)] = (__bf16)ra[i];
    lds_w[0][(e & 63) * 32 + (e >> 6)] = (__bf16)rw[i];
  }
  __syncthreads();

  v8f acc[4] = {};
  const int arow = wv * 16 + l16;
  const int akb  = lo ? 0 : 8;     // A-fragment K chunk base
  const int bkb  = lo ? 0 : 8;     // B-fragment K chunk base

  int buf = 0;
  for (int kk = 0; kk < D_MODEL; kk += 32) {
    const bool more = (kk + 32) < D_MODEL;
    // Issue next tile's global loads before touching the matrix pipe.
    if (more) {
#pragma unroll
      for (int i = 0; i < 16; ++i) {
        int e = t + i * 128;
        ra[i] = (float)A[(size_t)(m0 + (e >> 5)) * D_MODEL + kk + 32 + (e & 31)];
        rw[i] = W[(size_t)(kk + 32 + (e >> 6)) * D_MODEL + n0 + (e & 63)];
      }
    }

    v16bf afrag = join16(*(const v8bf*)&lds_a[buf][arow * 32 + akb],
                         *(const v8bf*)&lds_a[buf][arow * 32 + akb + 16]);
#pragma unroll
    for (int nb = 0; nb < 4; ++nb) {
      int col = nb * 16 + l16;
      v16bf bfrag = join16(*(const v8bf*)&lds_w[buf][col * 32 + bkb],
                           *(const v8bf*)&lds_w[buf][col * 32 + bkb + 16]);
      acc[nb] = __builtin_amdgcn_wmma_f32_16x16x32_bf16(
          false, afrag, false, bfrag, (short)0, acc[nb], false, false);
    }

    if (more) {
#pragma unroll
      for (int i = 0; i < 16; ++i) {
        int e = t + i * 128;
        lds_a[buf ^ 1][(e >> 5) * 32 + (e & 31)] = (__bf16)ra[i];
        lds_w[buf ^ 1][(e & 63) * 32 + (e >> 6)] = (__bf16)rw[i];
      }
    }
    __syncthreads();
    buf ^= 1;
  }

  // Epilogue: C layout -> VGPR v holds row (v | v+8), col = nb*16 + lane%16.
#pragma unroll
  for (int nb = 0; nb < 4; ++nb) {
    int n = n0 + nb * 16 + l16;
    float bv = bias[n];
#pragma unroll
    for (int v = 0; v < 8; ++v) {
      int mm = lo ? v : (v + 8);
      int m  = m0 + wv * 16 + mm;
      float val = acc[nb][v] + bv;
      if constexpr (MODE == 2) {
        ((float*)out)[(size_t)m * D_MODEL + n] = fmaxf(val, 0.0f);
      } else {
        int b = m >> 11, s = m & (S_LEN - 1);
        int h = n >> 6,  d = n & (DK - 1);
        __bf16* o = (__bf16*)out;
        if constexpr (MODE == 0)
          o[(((size_t)b * NHEAD + h) * S_LEN + s) * DK + d] = (__bf16)val;
        else
          o[(((size_t)b * NHEAD + h) * DK + d) * S_LEN + s] = (__bf16)val;
      }
    }
  }
}

// ---------------------------------------------------------------------------
// Flash attention: one wave per (b,h, 32-query tile) = two 16-query subtiles
// sharing each K/V fragment (doubles WMMA per byte of K/V traffic).
// S^T = K·Q^T so the score C-layout (keys in VGPRs, queries in lanes) is
// directly the A-layout for P in O = P @ V: exp/pack is lane-local.
// All K and V fragment loads are issued at the top of each 32-key step so V
// loads overlap the score WMMAs and the softmax VALU work.
// ---------------------------------------------------------------------------
__global__ __launch_bounds__(32) void attn_fwd(const __bf16* __restrict__ Qh,
                                               const __bf16* __restrict__ Kh,
                                               const __bf16* __restrict__ Vht,
                                               __bf16* __restrict__ Ohm) {
  const int lane  = threadIdx.x;
  const int l16   = lane & 15;
  const bool lo   = lane < 16;
  const int qbase = blockIdx.x * 32;
  const int bh    = blockIdx.y;
  // 1/sqrt(64) with log2(e) folded in; exp2f is a single v_exp_f32.
  const float sc  = 0.125f * 1.44269504088896341f;

  // Q^T B-fragments (dk is the WMMA K dim): lane = query column.
  const int qkb = lo ? 0 : 8;
  v16bf bq[2][2];
#pragma unroll
  for (int qt = 0; qt < 2; ++qt) {
    const __bf16* qrow = Qh + ((size_t)bh * S_LEN + qbase + qt * 16 + l16) * DK;
    bq[qt][0] = join16(*(const v8bf*)(qrow + qkb),      *(const v8bf*)(qrow + qkb + 16));
    bq[qt][1] = join16(*(const v8bf*)(qrow + 32 + qkb), *(const v8bf*)(qrow + 32 + qkb + 16));
  }

  v8f o[2][4] = {};
  float m_run[2] = {-__builtin_inff(), -__builtin_inff()};
  float l_run[2] = {0.0f, 0.0f};

  const __bf16* Kbase = Kh  + (size_t)bh * S_LEN * DK;
  const __bf16* Vbase = Vht + (size_t)bh * DK * S_LEN;

  for (int j = 0; j < S_LEN; j += 32) {
    // ---- issue ALL loads for this step up front ----
    v16bf ka[2][2];
#pragma unroll
    for (int kh = 0; kh < 2; ++kh) {
      const __bf16* krow = Kbase + (size_t)(j + kh * 16 + l16) * DK;
      const int c0 = lo ? 0 : 8;
      ka[kh][0] = join16(*(const v8bf*)(krow + c0),      *(const v8bf*)(krow + c0 + 16));
      ka[kh][1] = join16(*(const v8bf*)(krow + 32 + c0), *(const v8bf*)(krow + 32 + c0 + 16));
    }
    v16bf vb[4];
#pragma unroll
    for (int nb = 0; nb < 4; ++nb) {
      const __bf16* vrow = Vbase + (size_t)(nb * 16 + l16) * S_LEN + j + (lo ? 0 : 8);
      vb[nb] = join16(*(const v8bf*)vrow, *(const v8bf*)(vrow + 16));
    }
    if (j + 32 < S_LEN) {
      __builtin_prefetch(Kbase + (size_t)(j + 32 + lane) * DK, 0, 0);
      __builtin_prefetch(Vbase + (size_t)(lane * 2) * S_LEN + j + 32, 0, 0);
    }

    // ---- Scores^T for both query tiles (K fragments shared) ----
    v8f st[2][2];
#pragma unroll
    for (int qt = 0; qt < 2; ++qt) {
#pragma unroll
      for (int kh = 0; kh < 2; ++kh) {
        v8f s = {};
        s = __builtin_amdgcn_wmma_f32_16x16x32_bf16(false, ka[kh][0], false, bq[qt][0],
                                                    (short)0, s, false, false);
        s = __builtin_amdgcn_wmma_f32_16x16x32_bf16(false, ka[kh][1], false, bq[qt][1],
                                                    (short)0, s, false, false);
        st[qt][kh] = s;
      }
    }

    // ---- online softmax + PV per query tile ----
#pragma unroll
    for (int qt = 0; qt < 2; ++qt) {
      float mx = -__builtin_inff();
#pragma unroll
      for (int v = 0; v < 8; ++v) {
        mx = fmaxf(mx, st[qt][0][v] * sc);
        mx = fmaxf(mx, st[qt][1][v] * sc);
      }
      mx = fmaxf(mx, __shfl_xor(mx, 16, 32));
      float newm = fmaxf(m_run[qt], mx);

      float e0[8], e1[8], ls = 0.0f;
#pragma unroll
      for (int v = 0; v < 8; ++v) {
        e0[v] = exp2f(st[qt][0][v] * sc - newm);
        e1[v] = exp2f(st[qt][1][v] * sc - newm);
        ls += e0[v] + e1[v];
      }
      ls += __shfl_xor(ls, 16, 32);

      float fac = exp2f(m_run[qt] - newm);
      l_run[qt] = l_run[qt] * fac + ls;
      m_run[qt] = newm;

      // Rescale O: per-query factor lives in lanes; O rows live in VGPRs.
#pragma unroll
      for (int v = 0; v < 8; ++v) {
        float flo = __shfl(fac, v, 32);
        float fhi = __shfl(fac, v + 8, 32);
        float f = lo ? flo : fhi;
#pragma unroll
        for (int nb = 0; nb < 4; ++nb) o[qt][nb][v] *= f;
      }

      // Pack P as A-fragment (lane-local thanks to the S^T trick).
      v16bf pA;
#pragma unroll
      for (int i = 0; i < 8; ++i) { pA[i] = (__bf16)e0[i]; pA[i + 8] = (__bf16)e1[i]; }

#pragma unroll
      for (int nb = 0; nb < 4; ++nb)
        o[qt][nb] = __builtin_amdgcn_wmma_f32_16x16x32_bf16(
            false, pA, false, vb[nb], (short)0, o[qt][nb], false, false);
    }
  }

  // Normalize and store merged-head bf16 [B,S,D_MODEL].
  const int b = bh >> 4, h = bh & 15;
#pragma unroll
  for (int qt = 0; qt < 2; ++qt) {
    float rinv = 1.0f / l_run[qt];
#pragma unroll
    for (int v = 0; v < 8; ++v) {
      float flo = __shfl(rinv, v, 32);
      float fhi = __shfl(rinv, v + 8, 32);
      float f = lo ? flo : fhi;
      int mm = lo ? v : (v + 8);
      int qq = qbase + qt * 16 + mm;
#pragma unroll
      for (int nb = 0; nb < 4; ++nb) {
        int d = nb * 16 + l16;
        Ohm[((size_t)b * S_LEN + qq) * D_MODEL + h * DK + d] = (__bf16)(o[qt][nb][v] * f);
      }
    }
  }
}

// ---------------------------------------------------------------------------
extern "C" void kernel_launch(void* const* d_in, const int* in_sizes, int n_in,
                              void* d_out, int out_size, void* d_ws, size_t ws_size,
                              hipStream_t stream) {
  const float* q  = (const float*)d_in[0];
  const float* k  = (const float*)d_in[1];
  const float* v  = (const float*)d_in[2];
  const float* Wq = (const float*)d_in[3];
  const float* bq = (const float*)d_in[4];
  const float* Wk = (const float*)d_in[5];
  const float* bk = (const float*)d_in[6];
  const float* Wv = (const float*)d_in[7];
  const float* bv = (const float*)d_in[8];
  const float* Wo = (const float*)d_in[9];
  const float* bo = (const float*)d_in[10];

  const int B = in_sizes[0] / (S_LEN * D_MODEL);
  const int M = B * S_LEN;
  const size_t headBytes = (size_t)B * NHEAD * S_LEN * DK * sizeof(__bf16);

  char* ws = (char*)d_ws;
  __bf16* Qh  = (__bf16*)(ws);
  __bf16* Kh  = (__bf16*)(ws + headBytes);
  __bf16* Vt  = (__bf16*)(ws + 2 * headBytes);
  __bf16* Ohm = (__bf16*)(ws + 3 * headBytes);

  dim3 blk(128);
  dim3 grd(D_MODEL / 64, M / 64);
  proj_gemm<float, 0><<<grd, blk, 0, stream>>>(q, Wq, bq, Qh);
  proj_gemm<float, 0><<<grd, blk, 0, stream>>>(k, Wk, bk, Kh);
  proj_gemm<float, 1><<<grd, blk, 0, stream>>>(v, Wv, bv, Vt);
  attn_fwd<<<dim3(S_LEN / 32, B * NHEAD), dim3(32), 0, stream>>>(Qh, Kh, Vt, Ohm);
  proj_gemm<__bf16, 2><<<grd, blk, 0, stream>>>(Ohm, Wo, bo, d_out);
}
